// GIN_13941463843673
// MI455X (gfx1250) — matrix operand order repaired and checked by reading, not compiled, not run on hardware
//
#include <hip/hip_runtime.h>
#include <hip/hip_bf16.h>

#define NN 100000      // nodes
#define NE 1200000     // edges
#define DD 64          // feature dim
#define NL 3           // layers
#define NM 5000        // molecules
#define NO 24          // outputs
#define KMAX 192       // max GEMM K (JK: 3*64)

typedef __attribute__((ext_vector_type(16))) _Float16 v16h;
typedef __attribute__((ext_vector_type(8)))  _Float16 v8h;
typedef __attribute__((ext_vector_type(8)))  float    v8f;

union HFrag { v16h v; v8h h2[2]; _Float16 e[16]; };
union FFrag { v8f  v; float e[8]; };

// ---------------- small helper kernels ----------------

__global__ void k_zero(float* __restrict__ p, int n) {
    int i = blockIdx.x * blockDim.x + threadIdx.x;
    if (i < n) p[i] = 0.0f;
}

__global__ void k_embed(const int* __restrict__ atom_ids,
                        const float* __restrict__ embW,
                        float* __restrict__ x) {
    int i = blockIdx.x * blockDim.x + threadIdx.x;   // over NN*DD
    if (i < NN * DD) {
        int node = i >> 6, f = i & 63;
        x[i] = embW[atom_ids[node] * DD + f];
    }
}

// h[dst] += x[src] over edges (h was pre-initialized to x => h = x + agg)
__global__ void k_edge_agg(const int* __restrict__ src,
                           const int* __restrict__ dst,
                           const float* __restrict__ x,
                           float* __restrict__ h) {
    int i = blockIdx.x * blockDim.x + threadIdx.x;   // over NE*DD
    if (i < NE * DD) {
        int e = i >> 6, f = i & 63;
        atomicAdd(&h[dst[e] * DD + f], x[src[e] * DD + f]);
    }
}

// g[mol_ids[n]] += y[n]   (global_add_pool)
__global__ void k_pool(const int* __restrict__ mol_ids,
                       const float* __restrict__ y,
                       float* __restrict__ g) {
    int i = blockIdx.x * blockDim.x + threadIdx.x;   // over NN*DD
    if (i < NN * DD) {
        int node = i >> 6, f = i & 63;
        atomicAdd(&g[mol_ids[node] * DD + f], y[i]);
    }
}

// per-feature BN: scale = gamma*rsqrt(var+eps), shift = beta - mean*scale
__global__ void k_bn_finalize(const float* __restrict__ ssum,
                              const float* __restrict__ ssq,
                              const float* __restrict__ gamma,
                              const float* __restrict__ beta,
                              float invM,
                              float* __restrict__ scale,
                              float* __restrict__ shift) {
    int f = threadIdx.x;
    if (f < DD) {
        float mu  = ssum[f] * invM;
        float var = ssq[f] * invM - mu * mu;
        float s   = gamma[f] * rsqrtf(var + 1e-5f);
        scale[f] = s;
        shift[f] = beta[f] - mu * s;
    }
}

// ---------------- fused WMMA GEMM ----------------
// C[M,N] = act( f(A)[M,K] @ W[K,N] + bias ), f(A) = optional relu(A*scale+shift)
// optional second output C2 (strided, for JK concat slice)
// optional per-column stats: ssum[n] += sum_m C, ssq[n] += sum_m C^2
// block = 128 threads (4 waves); each block computes a 16-row strip,
// wave w computes N-tile w (N <= 64). K must be a multiple of 32, K <= KMAX.
__global__ __launch_bounds__(128)
void k_wmma_gemm(const float* __restrict__ A, int lda,
                 const float* __restrict__ scale, const float* __restrict__ shift,
                 int relu_a,
                 const float* __restrict__ W, const float* __restrict__ bias,
                 int M, int N, int K,
                 float* __restrict__ C, int ldc, int relu_c,
                 float* __restrict__ C2, int ldc2,
                 float* __restrict__ ssum, float* __restrict__ ssq) {
    // 32B-aligned so 128/256-bit LDS vector loads are legal; row strides
    // (KMAX=192 halves = 384B) preserve 32B alignment per row.
    __shared__ __align__(32) _Float16 sA [16][KMAX];   // A strip, f16
    __shared__ __align__(32) _Float16 sWt[DD][KMAX];   // W transposed [n][k], zero-padded to 64 cols

    const int tid  = threadIdx.x;
    const int row0 = blockIdx.x * 16;

    // Stage A strip (apply fused BN+ReLU transform), f32 -> f16
    for (int idx = tid; idx < 16 * K; idx += blockDim.x) {
        int r = idx / K, k = idx - r * K;
        int row = row0 + r;
        float v = (row < M) ? A[(long)row * lda + k] : 0.0f;
        if (scale) v = v * scale[k] + shift[k];
        if (relu_a) v = fmaxf(v, 0.0f);
        sA[r][k] = (_Float16)v;
    }
    // Stage weights transposed, zero-padded to 64 columns (kills per-element
    // bounds checks in the inner loop for the N=24 output projection)
    for (int idx = tid; idx < K * DD; idx += blockDim.x) {
        int k = idx >> 6, n = idx & 63;
        sWt[n][k] = (n < N) ? (_Float16)W[k * N + n] : (_Float16)0.0f;
    }
    __syncthreads();

    const int wave = tid >> 5;
    const int lane = tid & 31;
    const int m    = lane & 15;   // A-fragment row / B-fragment column (within tile)
    const int hi   = lane >> 4;
    const int ntiles = (N + 15) >> 4;

    for (int nt = wave; nt < ntiles; nt += 4) {
        const int n0  = nt << 4;
        const int col = n0 + m;
        FFrag acc; acc.v = (v8f){};

        const _Float16* rowA = &sA[m][0];
        const _Float16* colB = &sWt[col][0];

        for (int kb = 0; kb < K; kb += 32) {
            // A 16x32 f16 fragment (ISA 7.12.2): lane group hi reads two
            // contiguous 8-half runs: K = kb+8*hi+{0..7} and kb+16+8*hi+{0..7}
            HFrag a;
            a.h2[0] = *(const v8h*)(rowA + kb + (hi << 3));
            a.h2[1] = *(const v8h*)(rowA + kb + 16 + (hi << 3));
            // B 32x16 fragment: two contiguous 8-half runs per lane:
            // K = kb+8*hi+{0..7} and kb+16+8*hi+{0..7} at column `col` of the transposed tile
            HFrag b;
            b.h2[0] = *(const v8h*)(colB + kb + (hi << 3));
            b.h2[1] = *(const v8h*)(colB + kb + 16 + (hi << 3));

            acc.v = __builtin_amdgcn_wmma_f32_16x16x32_f16(
                false, a.v, false, b.v, (short)0, acc.v, false, false);
        }

        // Epilogue: C/D layout -> VGPR v holds row (v + 8*hi), col = n0 + (lane&15)
        const float bv = (col < N) ? bias[col] : 0.0f;
        float lsum = 0.0f, lsq = 0.0f;
#pragma unroll
        for (int v = 0; v < 8; ++v) {
            int row = row0 + v + (hi << 3);
            float val = acc.e[v] + bv;
            if (relu_c) val = fmaxf(val, 0.0f);
            if (row < M && col < N) {
                C[(long)row * ldc + col] = val;
                if (C2) C2[(long)row * ldc2 + col] = val;
                lsum += val;
                lsq  += val * val;
            }
        }
        if (ssum && col < N) {
            atomicAdd(&ssum[col], lsum);
            atomicAdd(&ssq[col],  lsq);
        }
    }
}

// ---------------- launcher ----------------

extern "C" void kernel_launch(void* const* d_in, const int* in_sizes, int n_in,
                              void* d_out, int out_size, void* d_ws, size_t ws_size,
                              hipStream_t stream) {
    const int*   atom_ids = (const int*)  d_in[0];
    const int*   edge     = (const int*)  d_in[1];
    const int*   mol_ids  = (const int*)  d_in[2];
    const float* emb_W    = (const float*)d_in[3];
    const float* gin_W1   = (const float*)d_in[4];
    const float* gin_b1   = (const float*)d_in[5];
    const float* gin_g1   = (const float*)d_in[6];
    const float* gin_be1  = (const float*)d_in[7];
    const float* gin_W2   = (const float*)d_in[8];
    const float* gin_b2   = (const float*)d_in[9];
    const float* jk_W     = (const float*)d_in[10];
    const float* jk_b     = (const float*)d_in[11];
    const float* ffn_W1   = (const float*)d_in[12];
    const float* ffn_b1   = (const float*)d_in[13];
    const float* ffn_g    = (const float*)d_in[14];
    const float* ffn_be   = (const float*)d_in[15];
    const float* ffn_W2   = (const float*)d_in[16];
    const float* ffn_b2   = (const float*)d_in[17];
    const float* out_W    = (const float*)d_in[18];
    const float* out_b    = (const float*)d_in[19];
    const int* src = edge;
    const int* dst = edge + NE;

    // workspace layout (bytes)
    const size_t ND  = (size_t)NN * DD * sizeof(float);    // 25.6 MB
    char* ws = (char*)d_ws;
    float* x    = (float*)(ws + 0);            // [NN,64]
    float* h    = (float*)(ws + ND);           // [NN,64]  (also y after layers)
    float* h1   = (float*)(ws + 2 * ND);       // [NN,64]  (also pool region after)
    float* xs   = (float*)(ws + 3 * ND);       // [NN,192] JK concat
    float* smem = (float*)(ws + 6 * ND);       // small: stats + scale/shift
    float* ssum  = smem;                       // [64]
    float* ssq   = smem + DD;                  // [64]
    float* scale = smem + 2 * DD;              // [64]
    float* shift = smem + 3 * DD;              // [64]
    // post-layer reuse
    float* y     = h;                          // [NN,64]
    float* gpool = h1;                         // [NM,64]
    float* t     = (float*)((char*)h1 + (size_t)NM * DD * sizeof(float));
    float* u     = (float*)((char*)h1 + 2 * (size_t)NM * DD * sizeof(float));

    const int TPB = 256;
    const int nodeElems = NN * DD;             // 6.4M
    const int edgeElems = NE * DD;             // 76.8M
    const int gridNodeGemm = NN / 16;          // 6250 (exact)
    const int gridMolGemm  = (NM + 15) / 16;   // 313

    // 1) embedding gather
    k_embed<<<(nodeElems + TPB - 1) / TPB, TPB, 0, stream>>>(atom_ids, emb_W, x);

    // 2) GIN layers
    for (int l = 0; l < NL; ++l) {
        // h = x; h[dst] += x[src]
        hipMemcpyAsync(h, x, ND, hipMemcpyDeviceToDevice, stream);
        k_edge_agg<<<(edgeElems + TPB - 1) / TPB, TPB, 0, stream>>>(src, dst, x, h);

        // h1 = h @ W1 + b1, fused per-column stats for BN
        k_zero<<<1, 128, 0, stream>>>(ssum, 2 * DD);
        k_wmma_gemm<<<gridNodeGemm, 128, 0, stream>>>(
            h, DD, nullptr, nullptr, 0,
            gin_W1 + (size_t)l * DD * DD, gin_b1 + l * DD,
            NN, DD, DD,
            h1, DD, /*relu_c=*/0, nullptr, 0, ssum, ssq);

        // BN scale/shift
        k_bn_finalize<<<1, DD, 0, stream>>>(ssum, ssq, gin_g1 + l * DD, gin_be1 + l * DD,
                                            1.0f / NN, scale, shift);

        // x = relu( relu(bn(h1)) @ W2 + b2 ); also write into xs[:, l*64:(l+1)*64]
        k_wmma_gemm<<<gridNodeGemm, 128, 0, stream>>>(
            h1, DD, scale, shift, /*relu_a=*/1,
            gin_W2 + (size_t)l * DD * DD, gin_b2 + l * DD,
            NN, DD, DD,
            x, DD, /*relu_c=*/1, xs + l * DD, NL * DD, nullptr, nullptr);
    }

    // 3) JK projection: y = xs @ jk_W + jk_b   (K = 192)
    k_wmma_gemm<<<gridNodeGemm, 128, 0, stream>>>(
        xs, NL * DD, nullptr, nullptr, 0,
        jk_W, jk_b,
        NN, DD, NL * DD,
        y, DD, 0, nullptr, 0, nullptr, nullptr);

    // 4) global_add_pool
    k_zero<<<(NM * DD + TPB - 1) / TPB, TPB, 0, stream>>>(gpool, NM * DD);
    k_pool<<<(nodeElems + TPB - 1) / TPB, TPB, 0, stream>>>(mol_ids, y, gpool);

    // 5) head MLP: t = g @ ffn_W1 + b1 (+stats) ; bn ; u = relu(bn(t)) @ ffn_W2 + b2
    k_zero<<<1, 128, 0, stream>>>(ssum, 2 * DD);
    k_wmma_gemm<<<gridMolGemm, 128, 0, stream>>>(
        gpool, DD, nullptr, nullptr, 0,
        ffn_W1, ffn_b1,
        NM, DD, DD,
        t, DD, 0, nullptr, 0, ssum, ssq);
    k_bn_finalize<<<1, DD, 0, stream>>>(ssum, ssq, ffn_g, ffn_be, 1.0f / NM, scale, shift);
    k_wmma_gemm<<<gridMolGemm, 128, 0, stream>>>(
        t, DD, scale, shift, 1,
        ffn_W2, ffn_b2,
        NM, DD, DD,
        u, DD, 0, nullptr, 0, nullptr, nullptr);

    // 6) output projection: out = u @ out_W + out_b   (N = 24)
    k_wmma_gemm<<<gridMolGemm, 128, 0, stream>>>(
        u, DD, nullptr, nullptr, 0,
        out_W, out_b,
        NM, NO, DD,
        (float*)d_out, NO, 0, nullptr, 0, nullptr, nullptr);
}
